// HeteroLinkPredictor_89000312308144
// MI455X (gfx1250) — compile-verified, matrix operand order checked by reading, not run on hardware
//
#include <hip/hip_runtime.h>
#include <hip/hip_bf16.h>
#include <math.h>

// ---------------------------------------------------------------------------
// HeteroLinkPredictor: score = sigmoid(W2 . relu(W1 . concat(ha[src],ht[dst]) + b1) + b2)
// bf16 WMMA (v_wmma_f32_16x16x32_bf16) with f32 accumulation on gfx1250.
// Double-buffered A/B fragments to decouple ds_load from v_wmma.
// ---------------------------------------------------------------------------

typedef __attribute__((ext_vector_type(16))) __bf16 v16bf;
typedef __attribute__((ext_vector_type(8)))  __bf16 v8bf;
typedef __attribute__((ext_vector_type(8)))  float  v8f;

#define HDIM     128            // hidden dim == per-table feature dim
#define TWOH     256            // concat feature dim
#define TILE_E   64             // edges per workgroup
#define SUB      (TILE_E / 16)  // 4 edge sub-tiles of 16
#define THREADS  256            // 8 wave32
#define WAVES    8
#define KSTEPS   (TWOH / 32)    // 8
#define FROWB    528            // LDS bytes per edge row: 256*2 + 16 pad (bank-conflict-free)

// ---- f32 -> bf16 conversion (ha / ht / W1 into workspace; all L2-resident) ----
__global__ void cvt_f32_to_bf16(const float* __restrict__ in,
                                __bf16* __restrict__ out, int n) {
  int i = blockIdx.x * blockDim.x + threadIdx.x;
  if (i < n) out[i] = (__bf16)in[i];
}

__device__ __forceinline__ v16bf load_frag(const __bf16* p) {
  v8bf lo = *(const v8bf*)(p);
  v8bf hi = *(const v8bf*)(p + 16);
  return __builtin_shufflevector(lo, hi,
      0,1,2,3,4,5,6,7,8,9,10,11,12,13,14,15);
}

__global__ __launch_bounds__(THREADS, 2)
void edge_mlp_wmma(const __bf16* __restrict__ haB, const __bf16* __restrict__ htB,
                   const __bf16* __restrict__ w1B,
                   const int* __restrict__ srcIdx, const int* __restrict__ dstIdx,
                   const float* __restrict__ b1, const float* __restrict__ w2,
                   const float* __restrict__ b2,
                   float* __restrict__ out, int nEdges)
{
  // 64 edge rows of 256 bf16, padded to 528B/row -> 33 KB; + 4 KB partials
  __shared__ __align__(16) char  featRaw[TILE_E * FROWB];
  __shared__ float plds[WAVES * 32 * SUB];

  const int tid      = threadIdx.x;
  const int tileBase = blockIdx.x * TILE_E;

  // ---------------- gather: concat(ha[src], ht[dst]) -> LDS (bf16) -----------
  // 32 x 16-byte chunks per edge (16 from ha row, 16 from ht row)
  for (int c = tid; c < TILE_E * 32; c += THREADS) {
    const int eL   = c >> 5;
    const int part = c & 31;
    const int e    = tileBase + eL;
    uint4 val = make_uint4(0u, 0u, 0u, 0u);
    if (e < nEdges) {
      if (part < 16) {
        const uint4* p = (const uint4*)(haB + (size_t)srcIdx[e] * HDIM);
        val = p[part];
      } else {
        const uint4* p = (const uint4*)(htB + (size_t)dstIdx[e] * HDIM);
        val = p[part - 16];
      }
    }
    *(uint4*)(featRaw + (size_t)eL * FROWB + (size_t)part * 16) = val;
  }
  __syncthreads();

  // ---------------- layer-1 GEMM tile via WMMA -------------------------------
  const int wave = tid >> 5;
  const int lane = tid & 31;
  const int half = lane >> 4;   // 0: lanes 0-15, 1: lanes 16-31
  const int l16  = lane & 15;

  v8f acc[SUB];
#pragma unroll
  for (int s = 0; s < SUB; ++s) acc[s] = (v8f){0.f,0.f,0.f,0.f,0.f,0.f,0.f,0.f};

  // A-matrix: W1 rows [16*wave, 16*wave+16).  Per ISA 16-bit A layout, lane l
  // holds K-chunks [k0 + 8*half, +8) and [k0 + 16 + 8*half, +8).
  const int cOff = 8 * half;
  const __bf16* w1row = w1B + (size_t)(16 * wave + l16) * TWOH + cOff;
  const __bf16* frow[SUB];
#pragma unroll
  for (int s = 0; s < SUB; ++s)
    frow[s] = (const __bf16*)(featRaw + (size_t)(s * 16 + l16) * FROWB) + cOff;

  // Double-buffered fragments: issue next K-step's loads before current WMMAs.
  v16bf Abuf[2];
  v16bf Bbuf[2][SUB];

  Abuf[0] = load_frag(w1row);
#pragma unroll
  for (int s = 0; s < SUB; ++s) Bbuf[0][s] = load_frag(frow[s]);

#pragma unroll
  for (int ki = 0; ki < KSTEPS; ++ki) {
    const int cur = ki & 1;
    const int nxt = cur ^ 1;
    if (ki + 1 < KSTEPS) {
      const int k1 = (ki + 1) * 32;
      Abuf[nxt] = load_frag(w1row + k1);
#pragma unroll
      for (int s = 0; s < SUB; ++s) Bbuf[nxt][s] = load_frag(frow[s] + k1);
    }
#pragma unroll
    for (int s = 0; s < SUB; ++s) {
      acc[s] = __builtin_amdgcn_wmma_f32_16x16x32_bf16(
          /*neg_a=*/false, Abuf[cur], /*neg_b=*/false, Bbuf[cur][s],
          /*c_mod=*/(short)0, acc[s], /*reuse_a=*/false, /*reuse_b=*/false);
    }
  }

  // ---------------- fused bias + ReLU + layer-2 dot --------------------------
  // C/D layout: VGPR v, lane l -> hid row m = 16*wave + v + 8*half, edge n = l16.
  const int mBase = 16 * wave + 8 * half;
  float4 b1lo = *(const float4*)(b1 + mBase);
  float4 b1hi = *(const float4*)(b1 + mBase + 4);
  float4 w2lo = *(const float4*)(w2 + mBase);
  float4 w2hi = *(const float4*)(w2 + mBase + 4);
  float bv[8] = {b1lo.x, b1lo.y, b1lo.z, b1lo.w, b1hi.x, b1hi.y, b1hi.z, b1hi.w};
  float wv[8] = {w2lo.x, w2lo.y, w2lo.z, w2lo.w, w2hi.x, w2hi.y, w2hi.z, w2hi.w};

#pragma unroll
  for (int s = 0; s < SUB; ++s) {
    float p = 0.f;
#pragma unroll
    for (int v = 0; v < 8; ++v) {
      float h = acc[s][v] + bv[v];
      h = fmaxf(h, 0.f);
      p = fmaf(h, wv[v], p);
    }
    plds[(wave * 32 + lane) * SUB + s] = p;
  }
  __syncthreads();

  // ---------------- deterministic cross-wave reduction + sigmoid -------------
  if (tid < TILE_E) {
    const int e = tileBase + tid;
    if (e < nEdges) {
      const int s = tid >> 4;
      const int n = tid & 15;
      float sum = b2[0];
#pragma unroll
      for (int w = 0; w < WAVES; ++w) {
        sum += plds[(w * 32 + n) * SUB + s];       // lanes 0-15  (m = v)
        sum += plds[(w * 32 + n + 16) * SUB + s];  // lanes 16-31 (m = v + 8)
      }
      out[e] = 1.0f / (1.0f + expf(-sum));
    }
  }
}

extern "C" void kernel_launch(void* const* d_in, const int* in_sizes, int n_in,
                              void* d_out, int out_size, void* d_ws, size_t ws_size,
                              hipStream_t stream) {
  const float* ha  = (const float*)d_in[0];
  const float* ht  = (const float*)d_in[1];
  const int*   src = (const int*)d_in[2];
  const int*   dst = (const int*)d_in[3];
  const float* W1w = (const float*)d_in[4];
  const float* W1b = (const float*)d_in[5];
  const float* W2w = (const float*)d_in[6];
  const float* W2b = (const float*)d_in[7];

  const int nHa = in_sizes[0];     // N_SRC * H
  const int nHt = in_sizes[1];     // N_DST * H
  const int nE  = in_sizes[2];     // N_EDGES
  const int nW1 = in_sizes[4];     // H * 2H

  // workspace layout (bf16 copies, 256B-aligned)
  size_t off = 0;
  __bf16* haB = (__bf16*)((char*)d_ws + off); off += ((size_t)nHa * 2 + 255) & ~(size_t)255;
  __bf16* htB = (__bf16*)((char*)d_ws + off); off += ((size_t)nHt * 2 + 255) & ~(size_t)255;
  __bf16* w1B = (__bf16*)((char*)d_ws + off); off += ((size_t)nW1 * 2 + 255) & ~(size_t)255;
  (void)ws_size; (void)n_in; (void)out_size;

  cvt_f32_to_bf16<<<(nHa + 255) / 256, 256, 0, stream>>>(ha,  haB, nHa);
  cvt_f32_to_bf16<<<(nHt + 255) / 256, 256, 0, stream>>>(ht,  htB, nHt);
  cvt_f32_to_bf16<<<(nW1 + 255) / 256, 256, 0, stream>>>(W1w, w1B, nW1);

  const int blocks = (nE + TILE_E - 1) / TILE_E;
  edge_mlp_wmma<<<blocks, THREADS, 0, stream>>>(
      haB, htB, w1B, src, dst, W1b, W2w, W2b, (float*)d_out, nE);
}
